// NeuralFingerprint_3616362463492
// MI455X (gfx1250) — matrix operand order repaired and checked by reading, not compiled, not run on hardware
//
#include <hip/hip_runtime.h>
#include <hip/hip_bf16.h>

#define F_DIM 64
#define L_DIM 1024
#define R_ROUNDS 3
#define NUM_CLASS 10

typedef __attribute__((ext_vector_type(16))) _Float16 v16h;
typedef __attribute__((ext_vector_type(8)))  float    v8f;

// fragment geometry: per (tile, h, lane) -> 16 contiguous f16 (32 bytes)
#define FRAG_HALVES 16
#define WO_TILES 64
#define WH_TILES 4
#define WO_PACK_HALVES (WO_TILES * 2 * 32 * FRAG_HALVES)   // 65536 per round
#define WH_PACK_HALVES (WH_TILES * 2 * 32 * FRAG_HALVES)   // 4096  per round

__device__ __forceinline__ unsigned pack_f16x2(float lo, float hi) {
  union { _Float16 h[2]; unsigned u; } v;
  v.h[0] = (_Float16)lo;
  v.h[1] = (_Float16)hi;
  return v.u;
}

// Pre-pack a row-major f32 weight W[64][ldb] into per-lane B-fragment order:
// halves index = ((tile*2 + h)*32 + lane)*16 + i,
// value = W[(32h + 16*(lane>>4) + i) * ldb + tile*16 + (lane&15)]
__global__ void pack_b_kernel(unsigned* __restrict__ dst,
                              const float* __restrict__ W, int ldb,
                              int nwords) {
  int idx = blockIdx.x * blockDim.x + threadIdx.x;
  if (idx >= nwords) return;
  int p    = idx & 7;           // dword within lane chunk (i = 2p, 2p+1)
  int lane = (idx >> 3) & 31;
  int h    = (idx >> 8) & 1;
  int t    = idx >> 9;
  int col  = t * 16 + (lane & 15);
  int k    = 32 * h + 16 * (lane >> 4) + 2 * p;
  dst[idx] = pack_f16x2(W[(size_t)k * ldb + col], W[(size_t)(k + 1) * ldb + col]);
}

// packed B fragment: one 32B vector per lane (2x global_load_b128)
__device__ __forceinline__ v16h load_b_frag(const _Float16* Wp, int tile,
                                            int h, int lane) {
  return *(const v16h*)(Wp + (((size_t)(tile * 2 + h) * 32 + lane) << 4));
}

// ---------------- small helpers --------------------------------------------
__global__ void zero_kernel(float* __restrict__ p, int n) {
  int i = blockIdx.x * blockDim.x + threadIdx.x;
  if (i < n) p[i] = 0.0f;
}

__global__ void gather_kernel(float* __restrict__ e,
                              const float* __restrict__ emb,
                              const int* __restrict__ nf, int n) {
  int idx = blockIdx.x * blockDim.x + threadIdx.x;   // n * 16 threads
  if (idx >= n * (F_DIM / 4)) return;
  int node = idx >> 4, c4 = idx & 15;
  const float4* src = (const float4*)(emb + (size_t)nf[node] * F_DIM);
  float4* dst = (float4*)(e + (size_t)node * F_DIM);
  dst[c4] = src[c4];
}

__global__ void scatter_kernel(float* __restrict__ nsum,
                               const float* __restrict__ e,
                               const int* __restrict__ esrc,
                               const int* __restrict__ edst, int ne) {
  int idx = blockIdx.x * blockDim.x + threadIdx.x;   // ne * 4 threads
  if (idx >= ne * 4) return;
  int edge = idx >> 2, q = idx & 3;
  int s = esrc[edge], d = edst[edge];
  const float4* sp = (const float4*)(e + (size_t)s * F_DIM + q * 16);
  float* dp = nsum + (size_t)d * F_DIM + q * 16;
#pragma unroll
  for (int j = 0; j < 4; ++j) {
    float4 v = sp[j];
    atomicAdd(dp + 4 * j + 0, v.x);
    atomicAdd(dp + 4 * j + 1, v.y);
    atomicAdd(dp + 4 * j + 2, v.z);
    atomicAdd(dp + 4 * j + 3, v.w);
  }
}

// A-fragment element mapping (16-bit A 16x32):
//   Afrag[h*512 + lane*16 + i] = A[m = lane&15][k = 32h + 16*(lane>>4) + i]
__device__ __forceinline__ void afrag_coords(int idx, int node0, int* node,
                                             int* k) {
  int i = idx & 15;
  int lane_ = (idx >> 4) & 31;
  int h = idx >> 9;
  int m = lane_ & 15, hh = lane_ >> 4;
  *k = 32 * h + 16 * hh + i;
  *node = node0 + m;
}

// ---------------- r = relu((e + nsum) @ Wh + bh) ---------------------------
__global__ void __launch_bounds__(128)
wh_kernel(float* __restrict__ r, const float* __restrict__ e,
          const float* __restrict__ nsum, const _Float16* __restrict__ WhP,
          const float* __restrict__ bh, int n) {
  __shared__ __align__(32) _Float16 Afrag[2 * 32 * 16];   // 2 KB
  int tid = threadIdx.x, lane = tid & 31, wave = tid >> 5;
  int node0 = blockIdx.x * 16;

  for (int idx = tid; idx < 1024; idx += 128) {
    int node, k;
    afrag_coords(idx, node0, &node, &k);
    float v = 0.0f;
    if (node < n)
      v = e[(size_t)node * F_DIM + k] + nsum[(size_t)node * F_DIM + k];
    Afrag[idx] = (_Float16)v;
  }
  __syncthreads();

  v16h a0 = *(const v16h*)(Afrag + lane * 16);
  v16h a1 = *(const v16h*)(Afrag + 512 + lane * 16);
  v16h b0 = load_b_frag(WhP, wave, 0, lane);
  v16h b1 = load_b_frag(WhP, wave, 1, lane);

  v8f acc = {};
  acc = __builtin_amdgcn_wmma_f32_16x16x32_f16(false, a0, false, b0,
                                               (short)0, acc, false, false);
  acc = __builtin_amdgcn_wmma_f32_16x16x32_f16(false, a1, false, b1,
                                               (short)0, acc, false, false);

  int hi = lane >> 4;
  int ncol = wave * 16 + (lane & 15);
  float bias = bh[ncol];
  float* rp = r + (size_t)(node0 + 8 * hi) * F_DIM + ncol;
  if (node0 + 16 <= n) {           // uniform fast path (n % 16 == 0 case)
#pragma unroll
    for (int g = 0; g < 8; ++g) {
      float v = acc[g] + bias;
      rp[(size_t)g * F_DIM] = v > 0.0f ? v : 0.0f;
    }
  } else {
#pragma unroll
    for (int g = 0; g < 8; ++g) {
      if (node0 + 8 * hi + g < n) {
        float v = acc[g] + bias;
        rp[(size_t)g * F_DIM] = v > 0.0f ? v : 0.0f;
      }
    }
  }
}

// ---------------- f += sum_nodes softmax(r @ Wo + bo) ----------------------
__global__ void __launch_bounds__(256)
fp_kernel(float* __restrict__ f, const float* __restrict__ r,
          const _Float16* __restrict__ WoP, const float* __restrict__ bo,
          int n) {
  __shared__ __align__(32) _Float16 Afrag[2 * 32 * 16];   // 2 KB
  __shared__ float wred[8][16];
  __shared__ float rowmax[16];
  __shared__ float rowinv[16];
  int tid = threadIdx.x, lane = tid & 31, wave = tid >> 5;
  int hi = lane >> 4;
  int node0 = blockIdx.x * 16;

  for (int idx = tid; idx < 1024; idx += 256) {
    int node, k;
    afrag_coords(idx, node0, &node, &k);
    float v = (node < n) ? r[(size_t)node * F_DIM + k] : 0.0f;
    Afrag[idx] = (_Float16)v;
  }
  __syncthreads();

  v16h a0 = *(const v16h*)(Afrag + lane * 16);
  v16h a1 = *(const v16h*)(Afrag + 512 + lane * 16);

  v8f acc[8];
#pragma unroll
  for (int j = 0; j < 8; ++j) {
    int tile = wave * 8 + j;
    v16h b0 = load_b_frag(WoP, tile, 0, lane);
    v16h b1 = load_b_frag(WoP, tile, 1, lane);
    v8f c = {};
    c = __builtin_amdgcn_wmma_f32_16x16x32_f16(false, a0, false, b0,
                                               (short)0, c, false, false);
    c = __builtin_amdgcn_wmma_f32_16x16x32_f16(false, a1, false, b1,
                                               (short)0, c, false, false);
    float bias = bo[tile * 16 + (lane & 15)];
#pragma unroll
    for (int g = 0; g < 8; ++g) acc[j][g] = c[g] + bias;
  }

  // row maxima: this lane holds rows m = g + 8*hi
  float lmax[8];
#pragma unroll
  for (int g = 0; g < 8; ++g) {
    float mx = acc[0][g];
#pragma unroll
    for (int j = 1; j < 8; ++j) mx = fmaxf(mx, acc[j][g]);
    lmax[g] = mx;
  }
#pragma unroll
  for (int s = 1; s < 16; s <<= 1) {
#pragma unroll
    for (int g = 0; g < 8; ++g)
      lmax[g] = fmaxf(lmax[g], __shfl_xor(lmax[g], s, 32));
  }
  if ((lane & 15) == 0) {
#pragma unroll
    for (int g = 0; g < 8; ++g) wred[wave][hi * 8 + g] = lmax[g];
  }
  __syncthreads();
  if (tid < 16) {
    float mx = wred[0][tid];
#pragma unroll
    for (int w = 1; w < 8; ++w) mx = fmaxf(mx, wred[w][tid]);
    rowmax[tid] = mx;
  }
  __syncthreads();

  float rmax[8];
#pragma unroll
  for (int g = 0; g < 8; ++g) rmax[g] = rowmax[g + 8 * hi];

  float lsum[8];
#pragma unroll
  for (int g = 0; g < 8; ++g) lsum[g] = 0.0f;
#pragma unroll
  for (int j = 0; j < 8; ++j) {
#pragma unroll
    for (int g = 0; g < 8; ++g) {
      float ev = __expf(acc[j][g] - rmax[g]);
      acc[j][g] = ev;
      lsum[g] += ev;
    }
  }
#pragma unroll
  for (int s = 1; s < 16; s <<= 1) {
#pragma unroll
    for (int g = 0; g < 8; ++g) lsum[g] += __shfl_xor(lsum[g], s, 32);
  }
  if ((lane & 15) == 0) {
#pragma unroll
    for (int g = 0; g < 8; ++g) wred[wave][hi * 8 + g] = lsum[g];
  }
  __syncthreads();
  if (tid < 16) {
    float s = wred[0][tid];
#pragma unroll
    for (int w = 1; w < 8; ++w) s += wred[w][tid];
    int node = node0 + tid;
    rowinv[tid] = (node < n && s > 0.0f) ? 1.0f / s : 0.0f;
  }
  __syncthreads();

  float rinv[8];
#pragma unroll
  for (int g = 0; g < 8; ++g) rinv[g] = rowinv[g + 8 * hi];

#pragma unroll
  for (int j = 0; j < 8; ++j) {
    float part = 0.0f;
#pragma unroll
    for (int g = 0; g < 8; ++g) part += acc[j][g] * rinv[g];
    int col = (wave * 8 + j) * 16 + (lane & 15);
    atomicAdd(&f[col], part);   // lanes n and n+16: rows 0-7 / 8-15 partials
  }
}

// ---------------- out = log_softmax(f @ Wcl + bcl) -------------------------
__global__ void __launch_bounds__(256)
cls_kernel(float* __restrict__ out, const float* __restrict__ f,
           const float* __restrict__ Wcl, const float* __restrict__ bcl) {
  __shared__ float red[256][NUM_CLASS];
  float p[NUM_CLASS];
#pragma unroll
  for (int c = 0; c < NUM_CLASS; ++c) p[c] = 0.0f;
  for (int k = threadIdx.x; k < L_DIM; k += 256) {
    float fk = f[k];
#pragma unroll
    for (int c = 0; c < NUM_CLASS; ++c) p[c] += fk * Wcl[k * NUM_CLASS + c];
  }
#pragma unroll
  for (int c = 0; c < NUM_CLASS; ++c) red[threadIdx.x][c] = p[c];
  __syncthreads();
  for (int s = 128; s > 0; s >>= 1) {
    if (threadIdx.x < (unsigned)s) {
#pragma unroll
      for (int c = 0; c < NUM_CLASS; ++c)
        red[threadIdx.x][c] += red[threadIdx.x + s][c];
    }
    __syncthreads();
  }
  if (threadIdx.x == 0) {
    float logits[NUM_CLASS];
    float mx = -3.4e38f;
#pragma unroll
    for (int c = 0; c < NUM_CLASS; ++c) {
      logits[c] = red[0][c] + bcl[c];
      mx = fmaxf(mx, logits[c]);
    }
    float se = 0.0f;
#pragma unroll
    for (int c = 0; c < NUM_CLASS; ++c) se += __expf(logits[c] - mx);
    float lse = __logf(se) + mx;
#pragma unroll
    for (int c = 0; c < NUM_CLASS; ++c) out[c] = logits[c] - lse;
  }
}

// ---------------- host-side orchestration ----------------------------------
extern "C" void kernel_launch(void* const* d_in, const int* in_sizes, int n_in,
                              void* d_out, int out_size, void* d_ws, size_t ws_size,
                              hipStream_t stream) {
  const int*   nf   = (const int*)d_in[0];
  const int*   esrc = (const int*)d_in[1];
  const int*   edst = (const int*)d_in[2];
  const float* emb  = (const float*)d_in[3];
  const float* Wh   = (const float*)d_in[4];
  const float* bh   = (const float*)d_in[5];
  const float* Wo   = (const float*)d_in[6];
  const float* bo   = (const float*)d_in[7];
  const float* Wcl  = (const float*)d_in[8];
  const float* bcl  = (const float*)d_in[9];
  float* out = (float*)d_out;

  int n  = in_sizes[0];
  int ne = in_sizes[1];

  float* e    = (float*)d_ws;
  float* r    = e    + (size_t)n * F_DIM;
  float* nsum = r    + (size_t)n * F_DIM;
  float* f    = nsum + (size_t)n * F_DIM;
  _Float16* WoP = (_Float16*)(f + L_DIM);                 // 3 * 128 KB
  _Float16* WhP = WoP + (size_t)R_ROUNDS * WO_PACK_HALVES; // 3 * 8 KB

  int nblk16 = (n + 15) / 16;

  // one-time (per launch) weight packing into f16 fragment layout
  for (int l = 0; l < R_ROUNDS; ++l) {
    pack_b_kernel<<<(WO_PACK_HALVES / 2 + 255) / 256, 256, 0, stream>>>(
        (unsigned*)(WoP + (size_t)l * WO_PACK_HALVES),
        Wo + (size_t)l * F_DIM * L_DIM, L_DIM, WO_PACK_HALVES / 2);
    pack_b_kernel<<<(WH_PACK_HALVES / 2 + 255) / 256, 256, 0, stream>>>(
        (unsigned*)(WhP + (size_t)l * WH_PACK_HALVES),
        Wh + (size_t)l * F_DIM * F_DIM, F_DIM, WH_PACK_HALVES / 2);
  }

  gather_kernel<<<(n * 16 + 255) / 256, 256, 0, stream>>>(e, emb, nf, n);
  zero_kernel<<<(L_DIM + 255) / 256, 256, 0, stream>>>(f, L_DIM);

  for (int l = 0; l < R_ROUNDS; ++l) {
    zero_kernel<<<(int)(((size_t)n * F_DIM + 255) / 256), 256, 0, stream>>>(
        nsum, n * F_DIM);
    scatter_kernel<<<(ne * 4 + 255) / 256, 256, 0, stream>>>(nsum, e, esrc,
                                                             edst, ne);
    wh_kernel<<<nblk16, 128, 0, stream>>>(r, e, nsum,
                                          WhP + (size_t)l * WH_PACK_HALVES,
                                          bh + l * F_DIM, n);
    fp_kernel<<<nblk16, 256, 0, stream>>>(f, r,
                                          WoP + (size_t)l * WO_PACK_HALVES,
                                          bo + l * L_DIM, n);
    float* tmp = e; e = r; r = tmp;
  }

  cls_kernel<<<1, 256, 0, stream>>>(out, f, Wcl, bcl);
}
